// MGKAttn_15891378995455
// MI455X (gfx1250) — matrix operand order repaired and checked by reading, not compiled, not run on hardware
//
#include <hip/hip_runtime.h>
#include <hip/hip_bf16.h>

typedef __attribute__((ext_vector_type(16))) _Float16 v16h;
typedef __attribute__((ext_vector_type(8)))  _Float16 v8h;
typedef __attribute__((ext_vector_type(8)))  float    v8f;
typedef __attribute__((ext_vector_type(4))) unsigned int u32x4;
typedef __attribute__((ext_vector_type(8))) int          i32x8;
typedef __attribute__((ext_vector_type(4))) int          i32x4;

#define S_LEN 1024
#define BATCH 8
#define DMODEL 512
#define NHEAD 8
#define DHEAD 64
#define MROWS (S_LEN*BATCH)   // 8192
#define SCALE 0.125f          // dh^-0.5

#if __has_builtin(__builtin_amdgcn_tensor_load_to_lds) && __has_builtin(__builtin_amdgcn_s_wait_tensorcnt)
#define HAVE_TDM 1
#else
#define HAVE_TDM 0
#endif

// ---------------- prep: f32 -> f16 convert ----------------
__global__ __launch_bounds__(256) void k_cvt(const float* __restrict__ src,
                                             _Float16* __restrict__ dst, int n) {
    int i = blockIdx.x * 256 + threadIdx.x;
    if (i < n) dst[i] = (_Float16)src[i];
}

// dst[nn][k] = src[k*ld + coloff + nn], dst is [512][512] f16 (B^T layout)
__global__ __launch_bounds__(256) void k_transpose(const float* __restrict__ src,
                                                   _Float16* __restrict__ dst,
                                                   int ld, int coloff) {
    int k  = blockIdx.x * 16 + threadIdx.x;
    int nn = blockIdx.y * 16 + threadIdx.y;
    dst[(size_t)nn * DMODEL + k] = (_Float16)src[(size_t)k * ld + coloff + nn];
}

// ---------------- QKV projection GEMM (WMMA f16) ----------------
// Per wave: 32x64 output tile -> 2 A-fragments x 4 B-fragments = 8 WMMA / k-step
__global__ __launch_bounds__(256) void k_gemm_qkv(const _Float16* __restrict__ A,
                                                  const _Float16* __restrict__ Bq,
                                                  const _Float16* __restrict__ Bk,
                                                  const _Float16* __restrict__ Bv,
                                                  _Float16* __restrict__ Qo,
                                                  _Float16* __restrict__ Ko,
                                                  _Float16* __restrict__ Vo) {
    const _Float16* Bt = (blockIdx.z == 0) ? Bq : (blockIdx.z == 1) ? Bk : Bv;
    _Float16*      Out = (blockIdx.z == 0) ? Qo : (blockIdx.z == 1) ? Ko : Vo;
    const int n0   = blockIdx.x * 64;
    const int m0   = blockIdx.y * 256 + (threadIdx.x >> 5) * 32;
    const int lane = threadIdx.x & 31;
    const int l    = lane & 15;
    const int hi   = lane >> 4;

    v8f acc[2][4] = {};
    for (int k0 = 0; k0 < DMODEL; k0 += 32) {
        const _Float16* pa = A + (size_t)(m0 + l) * DMODEL + k0 + hi * 16;
        __builtin_prefetch(pa + 64, 0, 3);
        union { v16h v; v8h h8[2]; } ua0, ua1;
        ua0.h8[0] = *(const v8h*)pa;
        ua0.h8[1] = *(const v8h*)(pa + 8);
        ua1.h8[0] = *(const v8h*)(pa + 16 * DMODEL);
        ua1.h8[1] = *(const v8h*)(pa + 16 * DMODEL + 8);
#pragma unroll
        for (int nt = 0; nt < 4; nt++) {
            const _Float16* pb = Bt + (size_t)(n0 + nt * 16 + l) * DMODEL + k0 + hi * 16;
            v16h bf = *(const v16h*)pb;
            acc[0][nt] = __builtin_amdgcn_wmma_f32_16x16x32_f16(
                false, ua0.v, false, bf, (short)0, acc[0][nt], false, false);
            acc[1][nt] = __builtin_amdgcn_wmma_f32_16x16x32_f16(
                false, ua1.v, false, bf, (short)0, acc[1][nt], false, false);
        }
    }
#pragma unroll
    for (int mh = 0; mh < 2; mh++)
#pragma unroll
        for (int nt = 0; nt < 4; nt++)
#pragma unroll
            for (int r = 0; r < 8; r++)
                Out[(size_t)(m0 + mh * 16 + r + hi * 8) * DMODEL + n0 + nt * 16 + l] =
                    (_Float16)acc[mh][nt][r];
}

// ---------------- per-row stats: q2, q.mu1, k2, |k-mu1|^2 ----------------
__global__ __launch_bounds__(256) void k_stats(const _Float16* __restrict__ Q,
                                               const _Float16* __restrict__ K,
                                               const float* __restrict__ mu,
                                               float* __restrict__ q2, float* __restrict__ qmu,
                                               float* __restrict__ k2, float* __restrict__ k2m) {
    int idx = blockIdx.x * 256 + threadIdx.x;   // (b*NHEAD+n)*S + s
    if (idx >= 64 * S_LEN) return;
    int s = idx & (S_LEN - 1);
    int bn = idx >> 10;
    int b = bn >> 3, n = bn & 7;
    const _Float16* qr = Q + ((size_t)s * BATCH + b) * DMODEL + n * DHEAD;
    const _Float16* kr = K + ((size_t)s * BATCH + b) * DMODEL + n * DHEAD;
    const float* m1 = mu + 512 + n * DHEAD;     // mu[1][n][:]
    float sq2 = 0, sqm = 0, sk2 = 0, sk2m = 0;
#pragma unroll 8
    for (int d = 0; d < DHEAD; d++) {
        float qv = (float)qr[d], kv = (float)kr[d], mv = m1[d];
        sq2 += qv * qv; sqm += qv * mv; sk2 += kv * kv;
        float km = kv - mv; sk2m += km * km;
    }
    q2[idx] = sq2; qmu[idx] = sqm; k2[idx] = sk2; k2m[idx] = sk2m;
}

// ---------------- fused two-kernel attention (flash style, no running max) ---
#define KSTR 96   // LDS row stride (halves): 64 data + 32 pad = 192B rows (32B aligned)
#define PSTR 48   // LDS row stride (halves) for P scratch (96B, 16B aligned)

__global__ __launch_bounds__(256) void k_attn(const _Float16* __restrict__ Q,
                                              const _Float16* __restrict__ K,
                                              const _Float16* __restrict__ V,
                                              const float* __restrict__ q2g,
                                              const float* __restrict__ qmug,
                                              const float* __restrict__ k2g,
                                              const float* __restrict__ k2mg,
                                              _Float16* __restrict__ vec) {
    __shared__ __align__(32) _Float16 Ksh[64 * KSTR];
    __shared__ __align__(32) _Float16 Vtsh[64 * KSTR];
    __shared__ __align__(32) _Float16 Psh[8][16 * PSTR];
    __shared__ float q2s[128], qmus[128], k2s[64], k2ms[64];

    const int bn = blockIdx.x;           // b*NHEAD + n
    const int b  = bn >> 3, n = bn & 7;
    const int i0 = blockIdx.y * 128;
    const int tid = threadIdx.x;
    const int w = tid >> 5, lane = tid & 31, l = lane & 15, hi = lane >> 4;
    const size_t statbase = (size_t)bn * S_LEN;

    if (tid < 128) { q2s[tid] = q2g[statbase + i0 + tid]; qmus[tid] = qmug[statbase + i0 + tid]; }

    // resident Q A-fragments for this wave's 16 rows (k = 0..31 and 32..63)
    const _Float16* qp = Q + ((size_t)(i0 + w * 16 + l) * BATCH + b) * DMODEL + n * DHEAD;
    union { v16h v; v8h h8[2]; } uq0, uq1;
    uq0.h8[0] = *(const v8h*)(qp + hi * 16);
    uq0.h8[1] = *(const v8h*)(qp + 8 + hi * 16);
    uq1.h8[0] = *(const v8h*)(qp + 32 + hi * 16);
    uq1.h8[1] = *(const v8h*)(qp + 40 + hi * 16);
    __syncthreads();

    float q2r[8], qmur[8];
#pragma unroll
    for (int r = 0; r < 8; r++) {
        q2r[r]  = q2s[w * 16 + hi * 8 + r];
        qmur[r] = qmus[w * 16 + hi * 8 + r];
    }

    v8f   oacc[4] = {};
    float lsum[8] = {};

    for (int jt = 0; jt < S_LEN; jt += 64) {
        __syncthreads();    // previous tile fully consumed
#if HAVE_TDM
        if (w == 0) {
            // TDM: DMA the 64x64 f16 K tile global->LDS, padding rows 128B->192B
            // (pad 16 DWORDs after every 32 DWORDs => LDS row stride = KSTR halves)
            unsigned lds_off = (unsigned)(size_t)&Ksh[0];
            unsigned long long ga =
                (unsigned long long)(size_t)(K + ((size_t)jt * BATCH + b) * DMODEL + n * DHEAD);
            u32x4 g0 = { 1u,                                  // count=1, user mode
                         lds_off,                             // lds_addr
                         (unsigned)(ga & 0xffffffffu),
                         (unsigned)((ga >> 32) & 0x01ffffffu) | (2u << 30) }; // type=2 (image)
            i32x8 g1 = { (1 << 16) | (1 << 20) | (4 << 22) | (15 << 25),
                         //  data_size=2B | pad_en | pad_interval=32dw | pad_amount=16dw
                         64 << 16,        // abar=0        | tensor_dim0 = 64
                         64 << 16,        // dim0_hi=0     | tensor_dim1 = 64
                         64 << 16,        // dim1_hi=0     | tile_dim0   = 64
                         64,              // tile_dim1=64  | tile_dim2   = 0
                         4096,            // tensor_dim0_stride = 4096 elements (= DMODEL*BATCH)
                         0, 0 };          // stride hi / tensor_dim1_stride (unused, 2D)
            i32x4 gz4 = { 0, 0, 0, 0 };
            i32x8 gz8 = { 0, 0, 0, 0, 0, 0, 0, 0 };
            __builtin_amdgcn_tensor_load_to_lds(g0, g1, gz4, gz4, gz8, 0);
            __builtin_amdgcn_s_wait_tensorcnt(0);
        }
        {   // V tile transposed into LDS (manual; TDM cannot transpose)
            int row = tid >> 2, part = tid & 3;
            const _Float16* vsrc = V + ((size_t)(jt + row) * BATCH + b) * DMODEL + n * DHEAD + part * 16;
#pragma unroll
            for (int e = 0; e < 16; e++) Vtsh[(part * 16 + e) * KSTR + row] = vsrc[e];
            if (tid < 64) { k2s[tid] = k2g[statbase + jt + tid]; k2ms[tid] = k2mg[statbase + jt + tid]; }
        }
#else
        {   // cooperative stage: K tile as-is, V tile transposed
            int row = tid >> 2, part = tid & 3;
            const _Float16* ksrc = K + ((size_t)(jt + row) * BATCH + b) * DMODEL + n * DHEAD + part * 16;
            *(v8h*)&Ksh[row * KSTR + part * 16]     = *(const v8h*)ksrc;
            *(v8h*)&Ksh[row * KSTR + part * 16 + 8] = *(const v8h*)(ksrc + 8);
            const _Float16* vsrc = V + ((size_t)(jt + row) * BATCH + b) * DMODEL + n * DHEAD + part * 16;
#pragma unroll
            for (int e = 0; e < 16; e++) Vtsh[(part * 16 + e) * KSTR + row] = vsrc[e];
            if (tid < 64) { k2s[tid] = k2g[statbase + jt + tid]; k2ms[tid] = k2mg[statbase + jt + tid]; }
        }
#endif
        __syncthreads();

#pragma unroll
        for (int jh = 0; jh < 2; jh++) {        // two 32-wide j windows
            const int js = jh * 32;
#pragma unroll
            for (int sub = 0; sub < 2; sub++) { // two 16-wide score tiles
                const int jr = js + sub * 16;
                v16h bf0 = *(const v16h*)&Ksh[(jr + l) * KSTR + hi * 16];       // dh 0..31
                v16h bf1 = *(const v16h*)&Ksh[(jr + l) * KSTR + 32 + hi * 16];  // dh 32..63
                v8f sacc = {};
                sacc = __builtin_amdgcn_wmma_f32_16x16x32_f16(false, uq0.v, false, bf0, (short)0, sacc, false, false);
                sacc = __builtin_amdgcn_wmma_f32_16x16x32_f16(false, uq1.v, false, bf1, (short)0, sacc, false, false);
                float k2v = k2s[jr + l], k2mv = k2ms[jr + l];
#pragma unroll
                for (int r = 0; r < 8; r++) {
                    float qk = sacc[r];
                    float t0 = q2r[r] + k2v  - 2.0f * qk;
                    float t1 = q2r[r] + k2mv - 2.0f * (qk - qmur[r]);
                    t0 = t0 > 0.0f ? t0 : 0.0f;
                    t1 = t1 > 0.0f ? t1 : 0.0f;
                    float sc = fmaxf(-0.5f * SCALE * t0, -1.5f * SCALE * t1);   // score <= 0
                    float p  = __expf(sc);                                      // in (0,1]
                    lsum[r] += p;
                    Psh[w][(r + hi * 8) * PSTR + sub * 16 + l] = (_Float16)p;
                }
            }
            asm volatile("s_wait_dscnt 0" ::: "memory");   // per-wave P writes visible
            union { v16h v; v8h h8[2]; } up;               // P as A-fragment (16 x 32 over j)
            up.h8[0] = *(const v8h*)&Psh[w][l * PSTR + hi * 16];
            up.h8[1] = *(const v8h*)&Psh[w][l * PSTR + 8 + hi * 16];
#pragma unroll
            for (int nt = 0; nt < 4; nt++) {
                v16h vf = *(const v16h*)&Vtsh[(nt * 16 + l) * KSTR + js + hi * 16];
                oacc[nt] = __builtin_amdgcn_wmma_f32_16x16x32_f16(false, up.v, false, vf, (short)0, oacc[nt], false, false);
            }
        }
    }

    // row-wise denominators: reduce across the 16-lane group, then normalize
#pragma unroll
    for (int r = 0; r < 8; r++) {
        float v = lsum[r];
#pragma unroll
        for (int off = 1; off < 16; off <<= 1) v += __shfl_xor(v, off, 32);
        lsum[r] = 1.0f / v;   // >= 1 always (diagonal term is exp(0))
    }
#pragma unroll
    for (int nt = 0; nt < 4; nt++)
#pragma unroll
        for (int r = 0; r < 8; r++) {
            int irow = i0 + w * 16 + hi * 8 + r;
            vec[((size_t)irow * BATCH + b) * DMODEL + n * DHEAD + nt * 16 + l] =
                (_Float16)(oacc[nt][r] * lsum[r]);
        }
}

// ---------------- output GEMM + residual: x = h + vec @ Wo ----------------
__global__ __launch_bounds__(256) void k_gemm_out(const _Float16* __restrict__ A,
                                                  const _Float16* __restrict__ Bt,
                                                  const float* __restrict__ h,
                                                  float* __restrict__ x) {
    const int n0   = blockIdx.x * 64;
    const int m0   = blockIdx.y * 256 + (threadIdx.x >> 5) * 32;
    const int lane = threadIdx.x & 31, l = lane & 15, hi = lane >> 4;
    v8f acc[2][4] = {};
    for (int k0 = 0; k0 < DMODEL; k0 += 32) {
        const _Float16* pa = A + (size_t)(m0 + l) * DMODEL + k0 + hi * 16;
        __builtin_prefetch(pa + 64, 0, 3);
        union { v16h v; v8h h8[2]; } ua0, ua1;
        ua0.h8[0] = *(const v8h*)pa;
        ua0.h8[1] = *(const v8h*)(pa + 8);
        ua1.h8[0] = *(const v8h*)(pa + 16 * DMODEL);
        ua1.h8[1] = *(const v8h*)(pa + 16 * DMODEL + 8);
#pragma unroll
        for (int nt = 0; nt < 4; nt++) {
            v16h bf = *(const v16h*)(Bt + (size_t)(n0 + nt * 16 + l) * DMODEL + k0 + hi * 16);
            acc[0][nt] = __builtin_amdgcn_wmma_f32_16x16x32_f16(false, ua0.v, false, bf, (short)0, acc[0][nt], false, false);
            acc[1][nt] = __builtin_amdgcn_wmma_f32_16x16x32_f16(false, ua1.v, false, bf, (short)0, acc[1][nt], false, false);
        }
    }
#pragma unroll
    for (int mh = 0; mh < 2; mh++)
#pragma unroll
        for (int nt = 0; nt < 4; nt++)
#pragma unroll
            for (int r = 0; r < 8; r++) {
                size_t idx = (size_t)(m0 + mh * 16 + r + hi * 8) * DMODEL + n0 + nt * 16 + l;
                x[idx] = acc[mh][nt][r] + h[idx];
            }
}

// ---------------- LayerNorm over D=512 per row ----------------
__global__ __launch_bounds__(256) void k_ln(const float* __restrict__ x,
                                            const float* __restrict__ gamma,
                                            const float* __restrict__ beta,
                                            float* __restrict__ out) {
    __shared__ float rbuf[8];
    __shared__ float stat[2];
    const int row = blockIdx.x, tid = threadIdx.x;
    const int lane = tid & 31, wid = tid >> 5;
    const float* xr = x + (size_t)row * DMODEL;
    float v0 = xr[tid], v1 = xr[tid + 256];
    float s = v0 + v1;
    for (int off = 16; off; off >>= 1) s += __shfl_xor(s, off, 32);
    if (lane == 0) rbuf[wid] = s;
    __syncthreads();
    if (tid == 0) { float t = 0; for (int i = 0; i < 8; i++) t += rbuf[i]; stat[0] = t * (1.0f / DMODEL); }
    __syncthreads();
    float mean = stat[0];
    float d0 = v0 - mean, d1 = v1 - mean;
    s = d0 * d0 + d1 * d1;
    for (int off = 16; off; off >>= 1) s += __shfl_xor(s, off, 32);
    __syncthreads();           // rbuf reuse
    if (lane == 0) rbuf[wid] = s;
    __syncthreads();
    if (tid == 0) { float t = 0; for (int i = 0; i < 8; i++) t += rbuf[i]; stat[1] = rsqrtf(t * (1.0f / DMODEL) + 1e-5f); }
    __syncthreads();
    float rstd = stat[1];
    out[(size_t)row * DMODEL + tid]       = d0 * rstd * gamma[tid] + beta[tid];
    out[(size_t)row * DMODEL + tid + 256] = d1 * rstd * gamma[tid + 256] + beta[tid + 256];
}

// ---------------- launch ----------------
extern "C" void kernel_launch(void* const* d_in, const int* in_sizes, int n_in,
                              void* d_out, int out_size, void* d_ws, size_t ws_size,
                              hipStream_t stream) {
    const float* h    = (const float*)d_in[0];   // [1024][8][512]
    const float* Wq   = (const float*)d_in[1];   // [512][512]
    const float* Wkv  = (const float*)d_in[2];   // [512][1024]
    const float* Wo   = (const float*)d_in[3];   // [512][512]
    const float* mu   = (const float*)d_in[4];   // [2][8][64]
    const float* gam  = (const float*)d_in[5];   // [512]
    const float* bet  = (const float*)d_in[6];   // [512]
    float* out = (float*)d_out;

    char* ws = (char*)d_ws;
    size_t off = 0;
    auto take = [&](size_t bytes) { char* p = ws + off; off = (off + bytes + 255) & ~(size_t)255; return p; };
    _Float16* h16   = (_Float16*)take((size_t)MROWS * DMODEL * 2);
    _Float16* Wq_t  = (_Float16*)take((size_t)DMODEL * DMODEL * 2);
    _Float16* Wk_t  = (_Float16*)take((size_t)DMODEL * DMODEL * 2);
    _Float16* Wv_t  = (_Float16*)take((size_t)DMODEL * DMODEL * 2);
    _Float16* Wo_t  = (_Float16*)take((size_t)DMODEL * DMODEL * 2);
    _Float16* Q16   = (_Float16*)take((size_t)MROWS * DMODEL * 2);
    _Float16* K16   = (_Float16*)take((size_t)MROWS * DMODEL * 2);
    _Float16* V16   = (_Float16*)take((size_t)MROWS * DMODEL * 2);
    _Float16* vec16 = (_Float16*)take((size_t)MROWS * DMODEL * 2);
    float* q2  = (float*)take((size_t)64 * S_LEN * 4);
    float* qmu = (float*)take((size_t)64 * S_LEN * 4);
    float* k2  = (float*)take((size_t)64 * S_LEN * 4);
    float* k2m = (float*)take((size_t)64 * S_LEN * 4);
    float* x   = (float*)take((size_t)MROWS * DMODEL * 4);

    // 1) precision/layout prep
    k_cvt<<<(MROWS * DMODEL + 255) / 256, 256, 0, stream>>>(h, h16, MROWS * DMODEL);
    dim3 tg(32, 32), tb(16, 16);
    k_transpose<<<tg, tb, 0, stream>>>(Wq,  Wq_t, 512, 0);
    k_transpose<<<tg, tb, 0, stream>>>(Wkv, Wk_t, 1024, 0);
    k_transpose<<<tg, tb, 0, stream>>>(Wkv, Wv_t, 1024, 512);
    k_transpose<<<tg, tb, 0, stream>>>(Wo,  Wo_t, 512, 0);
    // 2) QKV projection (WMMA, 32x64 per wave)
    k_gemm_qkv<<<dim3(8, 32, 3), 256, 0, stream>>>(h16, Wq_t, Wk_t, Wv_t, Q16, K16, V16);
    // 3) per-row kernel stats
    k_stats<<<(64 * S_LEN + 255) / 256, 256, 0, stream>>>(Q16, K16, mu, q2, qmu, k2, k2m);
    // 4) fused attention (WMMA flash, TDM-staged K tiles, no-max softmax)
    k_attn<<<dim3(64, 8), 256, 0, stream>>>(Q16, K16, V16, q2, qmu, k2, k2m, vec16);
    // 5) output projection + residual (WMMA)
    k_gemm_out<<<dim3(8, 32), 256, 0, stream>>>(vec16, Wo_t, h, x);
    // 6) LayerNorm
    k_ln<<<MROWS, 256, 0, stream>>>(x, gam, bet, out);
}
